// SharedScaleNA_28166395527734
// MI455X (gfx1250) — matrix operand order repaired and checked by reading, not compiled, not run on hardware
//
#include <hip/hip_runtime.h>
#include <hip/hip_bf16.h>
#include <stdint.h>

// ---------------------------------------------------------------------------
// SharedScaleNA for MI455X (gfx1250, wave32, WMMA)
//   phase 1: repack x (NCHW f32 -> NHWC bf16), w (OIHW f32 -> [tap][co][ci] bf16)
//   phase 2: implicit-GEMM 3x3 conv, double-buffered LDS, 2x2 WMMA tiles/wave,
//            branchless boundary masking
//   phase 3: per-pixel neighborhood attention (online softmax), f32 out
// ---------------------------------------------------------------------------

typedef __attribute__((ext_vector_type(16))) __bf16 v16bf;
typedef __attribute__((ext_vector_type(8)))  float  v8f;

#define IN_CH  192
#define W_IMG  96
#define HW     9216          // 96*96
#define COUT   768
#define NBATCH 8
#define LDSTR  40            // LDS row stride in halfwords (80B, 16B aligned)

struct U32x8 { uint4 lo, hi; };   // 32B container -> bit_cast to v16bf

__device__ __forceinline__ unsigned short f2bf(float f) {
  unsigned int u = __builtin_bit_cast(unsigned int, f);
  u += 0x7fffu + ((u >> 16) & 1u);          // round-to-nearest-even
  return (unsigned short)(u >> 16);
}
__device__ __forceinline__ float bflo(unsigned int u) { return __builtin_bit_cast(float, u << 16); }
__device__ __forceinline__ float bfhi(unsigned int u) { return __builtin_bit_cast(float, u & 0xffff0000u); }

// --------------------------- phase 1: repack -------------------------------

__global__ __launch_bounds__(256) void na_pack_x(const float* __restrict__ x,
                                                 unsigned short* __restrict__ xn) {
  unsigned idx = blockIdx.x * 256u + threadIdx.x;     // total 8*192*9216
  unsigned n   = idx / (IN_CH * HW);
  unsigned rem = idx % (IN_CH * HW);
  unsigned c   = rem / HW;
  unsigned p   = rem % HW;
  xn[(size_t)(n * HW + p) * IN_CH + c] = f2bf(x[idx]);
}

__global__ __launch_bounds__(256) void na_pack_w(const float* __restrict__ w,
                                                 unsigned short* __restrict__ wt) {
  unsigned idx = blockIdx.x * 256u + threadIdx.x;     // total 768*192*9
  unsigned o   = idx / (IN_CH * 9);
  unsigned r   = idx % (IN_CH * 9);
  unsigned c   = r / 9;
  unsigned tap = r % 9;
  wt[(size_t)(tap * COUT + o) * IN_CH + c] = f2bf(w[idx]);   // [tap][cout][cin]
}

// ---------------- phase 2: implicit-GEMM conv with WMMA --------------------
// block = 256 threads = 8 waves; macro-tile 64 couts x 128 pixels;
// wave = 32 couts x 32 pixels = 2x2 WMMA tiles; K = 9 taps * 6 chunks of 32.
// LDS double-buffered, one barrier per K-chunk, branchless halo masking.

__global__ __launch_bounds__(256) void na_conv_qkv(
    const unsigned short* __restrict__ xn,     // NHWC bf16
    const unsigned short* __restrict__ wt,     // [tap][co][ci] bf16
    const float* __restrict__ bias,
    unsigned short* __restrict__ qkv) {        // NHWC bf16 out
  __shared__ unsigned short ldsA[2][64  * LDSTR];   // [cout][k]
  __shared__ unsigned short ldsB[2][128 * LDSTR];   // [pixel][k]

  const int tid   = threadIdx.x;
  const int lane  = tid & 31;
  const int wave  = tid >> 5;
  const int half  = (lane >> 4) & 1;
  const int m16   = lane & 15;
  const int o_off = (wave & 1) * 32;           // wave's cout sub-tile (2x16)
  const int p_off = (wave >> 1) * 32;          // wave's pixel sub-tile (2x16)
  const int p0     = blockIdx.x * 128;         // 72 pixel tiles
  const int o_base = blockIdx.y * 64;          // 12 cout tiles
  const int nb     = blockIdx.z;               // batch

  // cooperative-load roles
  const int oa  = tid >> 2;                    // A: cout row 0..63
  const int kaq = (tid & 3) * 8;               // A: 8 channels
  const int pix = tid >> 1;                    // B: pixel row 0..127
  const int kbq = (tid & 1) * 16;              // B: 16 channels
  const int py  = (p0 + pix) / W_IMG;
  const int px  = (p0 + pix) % W_IMG;

  const unsigned short* wbase = wt + (size_t)(o_base + oa) * IN_CH + kaq;

  // fetch one K-chunk (ic = tap*6 + kc) into registers; branchless halo mask
  auto fetch = [&](int ic, uint4& rA, uint4& rB0, uint4& rB1, bool pf) {
    const int tap = ic / 6;
    const int c0  = (ic % 6) * 32;
    rA = *(const uint4*)(wbase + (size_t)tap * (COUT * IN_CH) + c0);
    const int ys = py + (tap / 3) - 1;
    const int xs = px + (tap % 3) - 1;
    const bool valid = (ys >= 0) & (ys < W_IMG) & (xs >= 0) & (xs < W_IMG);
    const int lin = valid ? (nb * HW + ys * W_IMG + xs) : 0;   // clamped: always safe
    const unsigned short* bsrc = xn + (size_t)lin * IN_CH + c0 + kbq;
    uint4 t0 = *(const uint4*)(bsrc);          // unconditional loads -> no EXEC branch
    uint4 t1 = *(const uint4*)(bsrc + 8);
    if (pf) __builtin_prefetch(bsrc + 32, 0, 1);   // next chunk, global_prefetch_b8
    const unsigned int msk = valid ? 0xffffffffu : 0u;   // v_cndmask selects
    rB0 = make_uint4(t0.x & msk, t0.y & msk, t0.z & msk, t0.w & msk);
    rB1 = make_uint4(t1.x & msk, t1.y & msk, t1.z & msk, t1.w & msk);
  };
  auto stage = [&](int buf, const uint4& rA, const uint4& rB0, const uint4& rB1) {
    *(uint4*)(&ldsA[buf][oa  * LDSTR + kaq]) = rA;
    *(uint4*)(&ldsB[buf][pix * LDSTR + kbq])     = rB0;
    *(uint4*)(&ldsB[buf][pix * LDSTR + kbq + 8]) = rB1;
  };

  v8f acc00 = {}, acc01 = {}, acc10 = {}, acc11 = {};

  {
    uint4 rA, rB0, rB1;
    fetch(0, rA, rB0, rB1, false);
    stage(0, rA, rB0, rB1);
  }
  __syncthreads();

  for (int ic = 0; ic < 54; ++ic) {
    uint4 rA, rB0, rB1;
    const bool have_next = (ic + 1) < 54;
    if (have_next) fetch(ic + 1, rA, rB0, rB1, true);

    const unsigned short* bufA = ldsA[ic & 1];
    const unsigned short* bufB = ldsB[ic & 1];
    // A fragments: lane row m; low half K{0..7,16..23}, high half K{8..15,24..31}
    U32x8 fa0, fa1, fb0, fb1;
    {
      const unsigned short* ra = bufA + (o_off + m16) * LDSTR + half * 8;
      fa0.lo = *(const uint4*)(ra);      fa0.hi = *(const uint4*)(ra + 16);
      fa1.lo = *(const uint4*)(ra + 16 * LDSTR);
      fa1.hi = *(const uint4*)(ra + 16 * LDSTR + 16);
    }
    // B fragments: lane col n; low half K{0..7,16..23}, high half K{8..15,24..31}
    {
      const unsigned short* rb = bufB + (p_off + m16) * LDSTR + half * 8;
      fb0.lo = *(const uint4*)(rb);      fb0.hi = *(const uint4*)(rb + 16);
      fb1.lo = *(const uint4*)(rb + 16 * LDSTR);
      fb1.hi = *(const uint4*)(rb + 16 * LDSTR + 16);
    }
    v16bf a0 = __builtin_bit_cast(v16bf, fa0);
    v16bf a1 = __builtin_bit_cast(v16bf, fa1);
    v16bf b0 = __builtin_bit_cast(v16bf, fb0);
    v16bf b1 = __builtin_bit_cast(v16bf, fb1);
    acc00 = __builtin_amdgcn_wmma_f32_16x16x32_bf16(false, a0, false, b0, (short)0, acc00, false, false);
    acc01 = __builtin_amdgcn_wmma_f32_16x16x32_bf16(false, a0, false, b1, (short)0, acc01, false, false);
    acc10 = __builtin_amdgcn_wmma_f32_16x16x32_bf16(false, a1, false, b0, (short)0, acc10, false, false);
    acc11 = __builtin_amdgcn_wmma_f32_16x16x32_bf16(false, a1, false, b1, (short)0, acc11, false, false);

    if (have_next) stage((ic + 1) & 1, rA, rB0, rB1);   // other buffer: safe, last read in prev iter, before its barrier
    __syncthreads();
  }

  // epilogue: D layout -> VGPR r: M = r + half*8, N = lane&15; bias + bf16 store
  #pragma unroll
  for (int ti = 0; ti < 2; ++ti) {
    #pragma unroll
    for (int tj = 0; tj < 2; ++tj) {
      const v8f& a = ti ? (tj ? acc11 : acc10) : (tj ? acc01 : acc00);
      const int pxl = p0 + p_off + tj * 16 + m16;
      const size_t rowbase = (size_t)(nb * HW + pxl) * COUT;
      #pragma unroll
      for (int r = 0; r < 8; ++r) {
        const int cout = o_base + o_off + ti * 16 + half * 8 + r;
        qkv[rowbase + cout] = f2bf(a[r] + bias[cout]);
      }
    }
  }
}

// ---------------- phase 3: neighborhood attention --------------------------
// one thread = one (pixel, batch, head); online softmax over ks*ks window.

__global__ __launch_bounds__(256) void na_attn(const unsigned short* __restrict__ qkv,
                                               float* __restrict__ out) {
  const int p  = blockIdx.x * 256 + threadIdx.x;   // 0..9215
  const int nb = blockIdx.y;
  const int h  = blockIdx.z;
  const int ks  = 3 + 2 * h;                       // 3,5,7,9
  const int pad = ks >> 1;
  const int y = p / W_IMG;
  const int x = p % W_IMG;

  const size_t base = (size_t)(nb * HW + p) * COUT + h * 192;
  unsigned int q[32];
  {
    const uint4* qp = (const uint4*)(qkv + base);   // 128B-aligned
    #pragma unroll
    for (int i = 0; i < 8; ++i) {
      uint4 t = qp[i];
      q[4*i+0] = t.x; q[4*i+1] = t.y; q[4*i+2] = t.z; q[4*i+3] = t.w;
    }
  }

  float m = -3.0e38f, l = 0.0f;
  float acc[64];
  #pragma unroll
  for (int d = 0; d < 64; ++d) acc[d] = 0.0f;

  for (int di = 0; di < ks; ++di) {
    const int yn = y + di - pad;
    if (yn < 0 || yn >= W_IMG) continue;
    for (int dj = 0; dj < ks; ++dj) {
      const int xw = x + dj - pad;
      if (xw < 0 || xw >= W_IMG) continue;
      const size_t nbase = (size_t)(nb * HW + yn * W_IMG + xw) * COUT + h * 192;
      const uint4* kp = (const uint4*)(qkv + nbase + 64);
      float s = 0.0f;
      #pragma unroll
      for (int i = 0; i < 8; ++i) {
        uint4 t = kp[i];
        s += bflo(t.x) * bflo(q[4*i+0]) + bfhi(t.x) * bfhi(q[4*i+0]);
        s += bflo(t.y) * bflo(q[4*i+1]) + bfhi(t.y) * bfhi(q[4*i+1]);
        s += bflo(t.z) * bflo(q[4*i+2]) + bfhi(t.z) * bfhi(q[4*i+2]);
        s += bflo(t.w) * bflo(q[4*i+3]) + bfhi(t.w) * bfhi(q[4*i+3]);
      }
      s *= 0.125f;                               // 1/sqrt(64)
      const float mn   = fmaxf(m, s);
      const float corr = __expf(m - mn);
      const float pw   = __expf(s - mn);
      l = l * corr + pw;
      const uint4* vp = (const uint4*)(qkv + nbase + 128);
      #pragma unroll
      for (int i = 0; i < 8; ++i) {
        uint4 t = vp[i];
        acc[8*i+0] = acc[8*i+0] * corr + pw * bflo(t.x);
        acc[8*i+1] = acc[8*i+1] * corr + pw * bfhi(t.x);
        acc[8*i+2] = acc[8*i+2] * corr + pw * bflo(t.y);
        acc[8*i+3] = acc[8*i+3] * corr + pw * bfhi(t.y);
        acc[8*i+4] = acc[8*i+4] * corr + pw * bflo(t.z);
        acc[8*i+5] = acc[8*i+5] * corr + pw * bfhi(t.z);
        acc[8*i+6] = acc[8*i+6] * corr + pw * bflo(t.w);
        acc[8*i+7] = acc[8*i+7] * corr + pw * bfhi(t.w);
      }
      m = mn;
    }
  }

  const float inv = 1.0f / l;
  const size_t ob = ((size_t)(h * NBATCH + nb) * 64) * HW + p;   // (head,N,D,H,W)
  #pragma unroll
  for (int d = 0; d < 64; ++d) out[ob + (size_t)d * HW] = acc[d] * inv;
}

// --------------------------------- launch ----------------------------------

extern "C" void kernel_launch(void* const* d_in, const int* in_sizes, int n_in,
                              void* d_out, int out_size, void* d_ws, size_t ws_size,
                              hipStream_t stream) {
  const float* x    = (const float*)d_in[0];   // (8,192,96,96)
  const float* wq   = (const float*)d_in[1];   // (768,192,3,3)
  const float* bq   = (const float*)d_in[2];   // (768,)
  float* out        = (float*)d_out;           // (4,8,64,96,96)

  // workspace layout (bytes): xnhwc bf16 | wt bf16 | qkv bf16
  unsigned short* xn  = (unsigned short*)d_ws;                           // 28,311,552 B
  unsigned short* wt  = (unsigned short*)((char*)d_ws + 28311552);       //  2,654,208 B
  unsigned short* qkv = (unsigned short*)((char*)d_ws + 30965760);       // 113,246,208 B

  na_pack_x<<<dim3(55296), dim3(256), 0, stream>>>(x, xn);
  na_pack_w<<<dim3(5184),  dim3(256), 0, stream>>>(wq, wt);
  na_conv_qkv<<<dim3(72, 12, 8), dim3(256), 0, stream>>>(xn, wt, bq, qkv);
  na_attn<<<dim3(36, 8, 4), dim3(256), 0, stream>>>(qkv, out);
}
